// MultiHeadAttention_36361193128313
// MI455X (gfx1250) — matrix operand order repaired and checked by reading, not compiled, not run on hardware
//
#include <hip/hip_runtime.h>
#include <hip/hip_bf16.h>
#include <math.h>

// ---------------------------------------------------------------------------
// Types
// ---------------------------------------------------------------------------
typedef __attribute__((ext_vector_type(16))) __bf16          v16bf;
typedef __attribute__((ext_vector_type(8)))  float           v8f;
typedef __attribute__((ext_vector_type(16))) unsigned short  v16us;
typedef __attribute__((ext_vector_type(8)))  unsigned short  v8us;

#define HD __device__ __forceinline__

static constexpr int Bb   = 2;
static constexpr int T    = 2048;
static constexpr int D    = 1024;
static constexpr int H    = 16;
static constexpr int DK   = 64;
static constexpr int TOPK = 64;
static constexpr int SSTR = 2048 + 4;   // fp32 score row stride (floats), bank stagger
static constexpr int ASTR = 2048 + 8;   // bf16 attn row stride (ushorts), keeps 16B align

// float -> bf16 round-to-nearest-even (bits)
HD unsigned short f2bf(float f) {
    unsigned int u = __float_as_uint(f);
    u += 0x7FFFu + ((u >> 16) & 1u);
    return (unsigned short)(u >> 16);
}

// Build a 16-element bf16 fragment from two 8-element contiguous chunks.
HD v16bf make_frag(const unsigned short* p0, const unsigned short* p1) {
    v8us a = *(const v8us*)p0;
    v8us b = *(const v8us*)p1;
    v16us t;
#pragma unroll
    for (int i = 0; i < 8; ++i) { t[i] = a[i]; t[i + 8] = b[i]; }
    return __builtin_bit_cast(v16bf, t);
}

HD v8f wmma_bf16(v16bf a, v16bf b, v8f c) {
    return __builtin_amdgcn_wmma_f32_16x16x32_bf16(
        /*neg_a=*/false, a, /*neg_b=*/false, b,
        /*c_mod=*/(short)0, c, /*reuse_a=*/false, /*reuse_b=*/false);
}

// monotonic float -> uint mapping (order preserving)
HD unsigned int sortable(float f) {
    unsigned int u = __float_as_uint(f);
    return u ^ ((u >> 31) ? 0xFFFFFFFFu : 0x80000000u);
}

// ---------------------------------------------------------------------------
// Kernel 1: fp32 -> bf16 conversion
// ---------------------------------------------------------------------------
__global__ void k_cvt_bf16(const float* __restrict__ x,
                           unsigned short* __restrict__ y, int n) {
    int i = blockIdx.x * blockDim.x + threadIdx.x;
    if (i < n) y[i] = f2bf(x[i]);
}

// ---------------------------------------------------------------------------
// Kernel 2: GEMM  Y(4096 x 1024) = Xbf(4096 x 1024) @ Wbf(1024 x 1024)^T + bias
// Each wave computes a 16x64 strip (4 accumulators, A fragment reused 4x),
// software-pipelined: next k-step fragments prefetched during current WMMAs.
// Block = 128 threads (4 waves) -> 16 rows x 256 cols.  grid = 1024.
//   mode 0: store bf16 into (b,h,t,dk) layout          (Q, K projections)
//   mode 1: store bf16 into (b,h,dk,t) layout (Vt)     (V projection)
//   mode 2: store fp32 row-major (B,T,D)               (output projection)
// ---------------------------------------------------------------------------
__global__ void k_proj(const unsigned short* __restrict__ X,
                       const unsigned short* __restrict__ W,
                       const float* __restrict__ bias,
                       void* __restrict__ out, int mode) {
    const int wave = threadIdx.x >> 5;
    const int lane = threadIdx.x & 31;
    const int mt   = blockIdx.x >> 2;            // 0..255 : 16-row tile
    const int cg   = blockIdx.x & 3;             // 0..3   : 256-col group
    const int nb   = cg * 256 + wave * 64;       // wave's 64-col strip base
    const int m    = lane & 15;
    const int half = lane >> 4;
    const int c0   = half * 8;                   // A fragment K-chunk base
    const int h0   = half * 8;                   // B fragment K-chunk base

    const unsigned short* xrow  = X + (size_t)(mt * 16 + m) * D;
    const unsigned short* wrow0 = W + (size_t)(nb +  0 + m) * D;
    const unsigned short* wrow1 = W + (size_t)(nb + 16 + m) * D;
    const unsigned short* wrow2 = W + (size_t)(nb + 32 + m) * D;
    const unsigned short* wrow3 = W + (size_t)(nb + 48 + m) * D;

    v16bf a  = make_frag(xrow  + c0, xrow  + c0 + 16);
    v16bf b0 = make_frag(wrow0 + h0, wrow0 + h0 + 16);
    v16bf b1 = make_frag(wrow1 + h0, wrow1 + h0 + 16);
    v16bf b2 = make_frag(wrow2 + h0, wrow2 + h0 + 16);
    v16bf b3 = make_frag(wrow3 + h0, wrow3 + h0 + 16);

    v8f acc0 = {}, acc1 = {}, acc2 = {}, acc3 = {};
    for (int k = 0; k < D; k += 32) {
        const int kn = (k + 32 < D) ? k + 32 : k;   // clamped prefetch (in-bounds)
        v16bf an  = make_frag(xrow  + kn + c0, xrow  + kn + c0 + 16);
        v16bf bn0 = make_frag(wrow0 + kn + h0, wrow0 + kn + h0 + 16);
        v16bf bn1 = make_frag(wrow1 + kn + h0, wrow1 + kn + h0 + 16);
        v16bf bn2 = make_frag(wrow2 + kn + h0, wrow2 + kn + h0 + 16);
        v16bf bn3 = make_frag(wrow3 + kn + h0, wrow3 + kn + h0 + 16);
        acc0 = wmma_bf16(a, b0, acc0);
        acc1 = wmma_bf16(a, b1, acc1);
        acc2 = wmma_bf16(a, b2, acc2);
        acc3 = wmma_bf16(a, b3, acc3);
        a = an; b0 = bn0; b1 = bn1; b2 = bn2; b3 = bn3;
    }

    const int n = lane & 15;
    v8f accs[4] = {acc0, acc1, acc2, acc3};
#pragma unroll
    for (int s = 0; s < 4; ++s) {
#pragma unroll
        for (int r = 0; r < 8; ++r) {
            const int mrow = half * 8 + r;
            const int grow = mt * 16 + mrow;      // [0, 4096)
            const int gcol = nb + s * 16 + n;     // [0, 1024)
            const float v  = accs[s][r] + bias[gcol];
            if (mode == 2) {
                ((float*)out)[(size_t)grow * D + gcol] = v;
            } else {
                const int bb = grow >> 11, t = grow & (T - 1);
                const int hh = gcol >> 6,  dk = gcol & (DK - 1);
                const unsigned short bv = f2bf(v);
                if (mode == 0)
                    ((unsigned short*)out)[(((size_t)(bb * H + hh) * T) + t) * DK + dk] = bv;
                else
                    ((unsigned short*)out)[(((size_t)(bb * H + hh) * DK) + dk) * T + t] = bv;
            }
        }
    }
}

// ---------------------------------------------------------------------------
// Kernel 3: attention core.  One block (128 thr / 4 waves) per
// (b, h, 16-query-row block).  grid = B*H*(T/16) = 4096.
// Phase 1: scores = Q Kt / 8 into LDS (fp32, exact accum), pipelined K loads
// Phase 2: exact 64th-largest per row via 4-bit radix select
// Phase 3: mask + Laplace CDF -> bf16 attn strip in LDS (A-frag layout)
// Phase 4: O = attn @ V via WMMA, pipelined (A = ds_load_b128, B from Vt)
// LDS: 16*SSTR fp32 + 16*ASTR bf16 + 16 u32  ~= 197 KB  (< 320 KB WGP LDS)
// ---------------------------------------------------------------------------
__global__ void k_attn(const unsigned short* __restrict__ Q,
                       const unsigned short* __restrict__ K,
                       const unsigned short* __restrict__ Vt,
                       const unsigned char* __restrict__ mask,
                       unsigned short* __restrict__ ctx) {
    extern __shared__ float sc[];                              // [16][SSTR] fp32
    unsigned short* attnb = (unsigned short*)(sc + 16 * SSTR); // [16][ASTR] bf16
    unsigned int*   thr   = (unsigned int*)(attnb + 16 * ASTR);

    const int wave = threadIdx.x >> 5;
    const int lane = threadIdx.x & 31;
    const int bh   = blockIdx.x >> 7;                   // 0..31
    const int qblk = blockIdx.x & 127;
    const int b    = bh >> 4;
    const int q0   = qblk * 16;
    const int m    = lane & 15;
    const int half = lane >> 4;
    const int c0   = half * 8;
    const int h0   = half * 8;

    // ---- Phase 1: score strip 16 x 2048 ----
    const unsigned short* qrow = Q + ((size_t)bh * T + q0 + m) * DK;
    const v16bf a0 = make_frag(qrow + c0,      qrow + c0 + 16);
    const v16bf a1 = make_frag(qrow + 32 + c0, qrow + 48 + c0);

    const unsigned short* kbase = K + ((size_t)bh * T + m) * DK;
    {
        const unsigned short* krow = kbase + (size_t)(wave * 16) * DK;
        v16bf b0 = make_frag(krow + h0,      krow + h0 + 16);
        v16bf b1 = make_frag(krow + 32 + h0, krow + 32 + h0 + 16);
        for (int kt = wave; kt < T / 16; kt += 4) {
            const int ktn = (kt + 4 < T / 16) ? kt + 4 : kt;   // clamped prefetch
            const unsigned short* krown = kbase + (size_t)(ktn * 16) * DK;
            v16bf nb0 = make_frag(krown + h0,      krown + h0 + 16);
            v16bf nb1 = make_frag(krown + 32 + h0, krown + 32 + h0 + 16);
            v8f c = {};
            c = wmma_bf16(a0, b0, c);
            c = wmma_bf16(a1, b1, c);
#pragma unroll
            for (int r = 0; r < 8; ++r)
                sc[(half * 8 + r) * SSTR + kt * 16 + m] = c[r] * 0.125f; // 1/sqrt(64)
            b0 = nb0; b1 = nb1;
        }
    }
    __syncthreads();

    // ---- Phase 2: per-row exact k-th largest (radix select on sortable bits)
    for (int row = wave; row < 16; row += 4) {
        const float* srow = sc + row * SSTR;
        unsigned int prefix = 0, remaining = TOPK;
        for (int pos = 28; pos >= 0; pos -= 4) {
            unsigned int cnt[16];
#pragma unroll
            for (int i = 0; i < 16; ++i) cnt[i] = 0;
            const unsigned int pmask = (pos == 28) ? 0u : (0xFFFFFFFFu << (pos + 4));
            for (int j = lane; j < T; j += 32) {
                unsigned int u = sortable(srow[j]);
                if ((u & pmask) == prefix) cnt[(u >> pos) & 15]++;
            }
#pragma unroll
            for (int i = 0; i < 16; ++i)
                for (int off = 16; off > 0; off >>= 1)
                    cnt[i] += (unsigned int)__shfl_xor((int)cnt[i], off, 32);
            unsigned int accu = 0, chosen = 0;
            for (int bkt = 15; bkt >= 0; --bkt) {
                if (accu + cnt[bkt] >= remaining) { chosen = (unsigned int)bkt; break; }
                accu += cnt[bkt];
            }
            prefix |= chosen << pos;
            remaining -= accu;
        }
        if (lane == 0) thr[row] = prefix;
    }
    __syncthreads();

    // ---- Phase 3: mask + Laplace CDF -> bf16 attn strip ----
    const unsigned char* mrow = mask + ((size_t)b * T + q0) * T;
    for (int idx = threadIdx.x; idx < 16 * T; idx += blockDim.x) {
        const int rr = idx >> 11, j = idx & (T - 1);
        const float s = sc[rr * SSTR + j];
        const bool keep = (sortable(s) >= thr[rr]) && !mrow[(size_t)rr * T + j];
        // laplace(x) = 0.5*(1+erf((x-mu)/(std*sqrt2))), mu=sqrt(.5), std=sqrt(.25*pi)
        const float a = keep
            ? 0.5f * (1.0f + erff((s - 0.70710678118654752f) * 0.79788456080286536f))
            : 0.0f;
        attnb[rr * ASTR + j] = f2bf(a);
    }
    __syncthreads();

    // ---- Phase 4: O(16 x 64) = attn(16 x 2048) @ V(2048 x 64), pipelined ----
    const int d0 = wave * 16;                            // each wave: one n-tile
    const unsigned short* vrow = Vt + ((size_t)bh * DK + d0 + m) * T;
    const unsigned short* arow = attnb + m * ASTR;
    v16bf af = make_frag(arow + c0, arow + c0 + 16);
    v16bf bf = make_frag(vrow + h0, vrow + h0 + 16);
    v8f acc = {};
    for (int kt = 0; kt < T / 32; ++kt) {
        const int kbn = ((kt + 1 < T / 32) ? kt + 1 : kt) * 32;  // clamped prefetch
        v16bf an = make_frag(arow + kbn + c0, arow + kbn + c0 + 16);
        v16bf bn = make_frag(vrow + kbn + h0, vrow + kbn + h0 + 16);
        acc = wmma_bf16(af, bf, acc);
        af = an; bf = bn;
    }
    const int h = bh & 15;
#pragma unroll
    for (int r = 0; r < 8; ++r) {
        const int mrow2 = half * 8 + r;
        ctx[((size_t)b * T + q0 + mrow2) * D + h * DK + d0 + (lane & 15)] = f2bf(acc[r]);
    }
}

// ---------------------------------------------------------------------------
// Host launcher
// ---------------------------------------------------------------------------
extern "C" void kernel_launch(void* const* d_in, const int* in_sizes, int n_in,
                              void* d_out, int out_size, void* d_ws, size_t ws_size,
                              hipStream_t stream) {
    (void)in_sizes; (void)n_in; (void)out_size; (void)ws_size;

    const float* query = (const float*)d_in[0];
    const float* key   = (const float*)d_in[1];
    const float* value = (const float*)d_in[2];
    const unsigned char* mask = (const unsigned char*)d_in[3];
    const float* wq = (const float*)d_in[4];
    const float* bq = (const float*)d_in[5];
    const float* wk = (const float*)d_in[6];
    const float* bk = (const float*)d_in[7];
    const float* wv = (const float*)d_in[8];
    const float* bv = (const float*)d_in[9];
    const float* wo = (const float*)d_in[10];
    const float* bo = (const float*)d_in[11];

    const int NX = Bb * T * D;   // 4 194 304  (also B*H*T*DK)
    const int NW = D * D;        // 1 048 576

    char* ws = (char*)d_ws;
    size_t off = 0;
    auto alloc = [&](size_t bytes) {
        void* p = ws + off;
        off += (bytes + 255) & ~(size_t)255;
        return p;
    };
    unsigned short* xq  = (unsigned short*)alloc((size_t)NX * 2);
    unsigned short* xk  = (unsigned short*)alloc((size_t)NX * 2);
    unsigned short* xv  = (unsigned short*)alloc((size_t)NX * 2);
    unsigned short* wqb = (unsigned short*)alloc((size_t)NW * 2);
    unsigned short* wkb = (unsigned short*)alloc((size_t)NW * 2);
    unsigned short* wvb = (unsigned short*)alloc((size_t)NW * 2);
    unsigned short* wob = (unsigned short*)alloc((size_t)NW * 2);
    unsigned short* Qb  = (unsigned short*)alloc((size_t)NX * 2);  // (b,h,t,dk)
    unsigned short* Kb  = (unsigned short*)alloc((size_t)NX * 2);  // (b,h,t,dk)
    unsigned short* Vtb = (unsigned short*)alloc((size_t)NX * 2);  // (b,h,dk,t)
    unsigned short* ctx = (unsigned short*)alloc((size_t)NX * 2);  // (b,t,h*dk)

    const int CT = 256;
    // 1) fp32 -> bf16 conversions
    k_cvt_bf16<<<(NX + CT - 1) / CT, CT, 0, stream>>>(query, xq, NX);
    k_cvt_bf16<<<(NX + CT - 1) / CT, CT, 0, stream>>>(key,   xk, NX);
    k_cvt_bf16<<<(NX + CT - 1) / CT, CT, 0, stream>>>(value, xv, NX);
    k_cvt_bf16<<<(NW + CT - 1) / CT, CT, 0, stream>>>(wq, wqb, NW);
    k_cvt_bf16<<<(NW + CT - 1) / CT, CT, 0, stream>>>(wk, wkb, NW);
    k_cvt_bf16<<<(NW + CT - 1) / CT, CT, 0, stream>>>(wv, wvb, NW);
    k_cvt_bf16<<<(NW + CT - 1) / CT, CT, 0, stream>>>(wo, wob, NW);

    // 2) projections (WMMA GEMMs): grid = (4096/16)*(1024/256) = 1024
    const int gemmGrid = (Bb * T / 16) * (D / 256);
    k_proj<<<gemmGrid, 128, 0, stream>>>(xq, wqb, bq, (void*)Qb,  0);
    k_proj<<<gemmGrid, 128, 0, stream>>>(xk, wkb, bk, (void*)Kb,  0);
    k_proj<<<gemmGrid, 128, 0, stream>>>(xv, wvb, bv, (void*)Vtb, 1);

    // 3) attention core (scores LDS strip + radix top-k + laplace + attn@V)
    const size_t ldsBytes = (size_t)(16 * SSTR) * sizeof(float)
                          + (size_t)(16 * ASTR) * sizeof(unsigned short)
                          + 16 * sizeof(unsigned int);
    hipFuncSetAttribute((const void*)k_attn,
                        hipFuncAttributeMaxDynamicSharedMemorySize, (int)ldsBytes);
    k_attn<<<Bb * H * (T / 16), 128, ldsBytes, stream>>>(Qb, Kb, Vtb, mask, ctx);

    // 4) output projection -> fp32
    k_proj<<<gemmGrid, 128, 0, stream>>>(ctx, wob, bo, d_out, 2);
}
